// BigBirdStyleAttention_17927193493836
// MI455X (gfx1250) — matrix operand order repaired and checked by reading, not compiled, not run on hardware
//
#include <hip/hip_runtime.h>

// ---------------------------------------------------------------------------
// BigBird-style attention for MI455X (gfx1250, wave32, WMMA bf16 16x16x32).
// - All heavy math through v_wmma_f32_16x16x32_bf16 (fp32 accumulate).
// - Block-sparse flash attention skips dead 64x64 key blocks (exact).
// - Mask pre-packed to 64-bit column bitmaps (8 b64 loads/chunk vs 32 u8).
// - Double-buffered GEMM with GLOBAL_LOAD_ASYNC_TO_LDS_B128 when available.
// ---------------------------------------------------------------------------

#define S_LEN   2048
#define HIDDEN  2048
#define NHEAD   16
#define DHEAD   128
#define NBATCH  2
#define BLKSZ   64
#define NBLK    32                   // S_LEN / BLKSZ
#define QKSCALE 0.08838834764831845f // 128^-0.5
#define NEGBIG  (-1e30f)

#if defined(__has_builtin)
#if __has_builtin(__builtin_amdgcn_global_load_async_to_lds_b128) && \
    __has_builtin(__builtin_amdgcn_s_wait_asynccnt)
#define BB_ASYNC 1
#else
#define BB_ASYNC 0
#endif
#else
#define BB_ASYNC 0
#endif

#if BB_ASYNC
// builtin signature (from compiler diagnostics): arg0 = v4i* (global source),
// arg1 = LDS destination, arg2/arg3 = imm offset / cpol.
typedef __attribute__((__vector_size__(4 * sizeof(int)))) int bb_v4i;
#define AS1P(p) ((bb_v4i*)(p))
#define AS3P(p) ((__attribute__((address_space(3))) bb_v4i*)(p))
#endif

typedef __attribute__((ext_vector_type(16))) __bf16 v16bf;
typedef __attribute__((ext_vector_type(8)))  float  v8f;

union FragA {          // one WMMA 16-bit A/B operand: 16 bf16 = 32 bytes
  v16bf v;
  uint4 u[2];
};

__device__ __forceinline__ unsigned short f2bf(float x) {
  unsigned u = __float_as_uint(x);
  unsigned r = u + 0x7FFFu + ((u >> 16) & 1u);   // round-to-nearest-even
  return (unsigned short)(r >> 16);
}

// ---------------------------------------------------------------------------
// Kernel 1: fp32 -> bf16 bulk convert (X), 4 elems/thread.
// ---------------------------------------------------------------------------
__global__ __launch_bounds__(256) void bb_convert(
    const float* __restrict__ src, unsigned short* __restrict__ dst, int n4) {
  int i = blockIdx.x * 256 + threadIdx.x;
  if (i >= n4) return;
  float4 f = ((const float4*)src)[i];
  uint2 p;
  p.x = (unsigned)f2bf(f.x) | ((unsigned)f2bf(f.y) << 16);
  p.y = (unsigned)f2bf(f.z) | ((unsigned)f2bf(f.w) << 16);
  ((uint2*)dst)[i] = p;
}

// ---------------------------------------------------------------------------
// Kernel 2: pack mask bytes into u64 column bitmaps: bit c of
// maskBits[row*NBLK + kb] = mask[row][kb*64 + c].
// ---------------------------------------------------------------------------
__global__ __launch_bounds__(256) void bb_mask_pack(
    const unsigned char* __restrict__ mask, unsigned long long* __restrict__ bits) {
  int gid = blockIdx.x * 256 + threadIdx.x;    // 2048*32 words
  int row = gid >> 5, kb = gid & 31;
  const unsigned* p = (const unsigned*)(mask + (size_t)row * S_LEN + kb * BLKSZ);
  unsigned long long w = 0;
  for (int i = 0; i < 16; ++i) {
    unsigned x = p[i];
    if (x & 0x000000FFu) w |= 1ull << (4 * i + 0);
    if (x & 0x0000FF00u) w |= 1ull << (4 * i + 1);
    if (x & 0x00FF0000u) w |= 1ull << (4 * i + 2);
    if (x & 0xFF000000u) w |= 1ull << (4 * i + 3);
  }
  bits[gid] = w;
}

// ---------------------------------------------------------------------------
// Kernel 3: per-64x64-block "any live" flag (from the packed bitmaps).
// grid = NBLK*NBLK, 64 threads: each thread ORs one row's word.
// ---------------------------------------------------------------------------
__global__ __launch_bounds__(64) void bb_mask_blocks(
    const unsigned long long* __restrict__ bits, unsigned char* __restrict__ blockAny) {
  __shared__ int flag;
  if (threadIdx.x == 0) flag = 0;
  __syncthreads();
  int qb = blockIdx.x % NBLK, kb = blockIdx.x / NBLK;
  unsigned long long w = bits[(size_t)(qb * BLKSZ + threadIdx.x) * NBLK + kb];
  if (w) atomicOr(&flag, 1);
  __syncthreads();
  if (threadIdx.x == 0) blockAny[qb * NBLK + kb] = (unsigned char)flag;
}

// ---------------------------------------------------------------------------
// Kernel 4: bf16 WMMA GEMM  C(MxN) = A(MxK)bf16 @ B(KxN)f32->bf16.
// Block tile 128x128, K-step 32, 8 waves (2x4), wave tile 64x32.
// Double-buffered LDS; A tile copied with async global->LDS when available;
// B tile staged transposed (n-major) so B-operand lane reads are contiguous.
// ---------------------------------------------------------------------------
template <bool OUT_BF16>
__global__ __launch_bounds__(256) void bb_gemm(
    const unsigned short* __restrict__ A, const float* __restrict__ B,
    void* __restrict__ Cptr, int M, int N, int K) {
  __shared__ unsigned short ldsA[2][128 * 32];   // m-major, stride 32
  __shared__ unsigned short ldsBt[2][128 * 32];  // n-major, stride 32

  const int tid = threadIdx.x;
  const int lane = tid & 31, half = lane >> 4, lane16 = lane & 15;
  const int wid = tid >> 5;
  const int wm = wid & 1, wn = wid >> 1;
  const int m0 = blockIdx.y * 128, n0 = blockIdx.x * 128;

  // this thread's two A chunks (8 bf16 each) within the 128x32 tile
  const int ar0 = tid >> 2, ak0 = (tid & 3) * 8;
  const int ar1 = (tid + 256) >> 2, ak1 = ((tid + 256) & 3) * 8;

  v8f acc[4][2];
  for (int ms = 0; ms < 4; ++ms)
    for (int ns = 0; ns < 2; ++ns)
      for (int i = 0; i < 8; ++i) acc[ms][ns][i] = 0.f;

  float4 bregs[4];
#if !BB_ASYNC
  uint4 aregs[2];
#endif

  auto loadB = [&](int k0) {
    for (int i = 0; i < 4; ++i) {
      int g = tid + i * 256;
      int kk = g >> 5, nof = (g & 31) * 4;
      bregs[i] = *(const float4*)&B[(size_t)(k0 + kk) * N + n0 + nof];
    }
  };
  auto issueA = [&](int buf, int k0) {
#if BB_ASYNC
    __builtin_amdgcn_global_load_async_to_lds_b128(
        AS1P(&A[(size_t)(m0 + ar0) * K + k0 + ak0]),
        AS3P(&ldsA[buf][ar0 * 32 + ak0]), 0, 0);
    __builtin_amdgcn_global_load_async_to_lds_b128(
        AS1P(&A[(size_t)(m0 + ar1) * K + k0 + ak1]),
        AS3P(&ldsA[buf][ar1 * 32 + ak1]), 0, 0);
#else
    aregs[0] = *(const uint4*)&A[(size_t)(m0 + ar0) * K + k0 + ak0];
    aregs[1] = *(const uint4*)&A[(size_t)(m0 + ar1) * K + k0 + ak1];
    (void)buf;
#endif
  };
  auto commit = [&](int buf) {
#if BB_ASYNC
    __builtin_amdgcn_s_wait_asynccnt(0);
#else
    *(uint4*)&ldsA[buf][ar0 * 32 + ak0] = aregs[0];
    *(uint4*)&ldsA[buf][ar1 * 32 + ak1] = aregs[1];
#endif
    for (int i = 0; i < 4; ++i) {
      int g = tid + i * 256;
      int kk = g >> 5, nof = (g & 31) * 4;
      float4 f = bregs[i];
      ldsBt[buf][(nof + 0) * 32 + kk] = f2bf(f.x);
      ldsBt[buf][(nof + 1) * 32 + kk] = f2bf(f.y);
      ldsBt[buf][(nof + 2) * 32 + kk] = f2bf(f.z);
      ldsBt[buf][(nof + 3) * 32 + kk] = f2bf(f.w);
    }
  };
  auto compute = [&](int buf) {
    FragA af[4], bf[2];
    for (int ms = 0; ms < 4; ++ms) {
      int r = wm * 64 + ms * 16 + lane16;        // A layout: K = 8*half + e
      af[ms].u[0] = *(const uint4*)&ldsA[buf][r * 32 + 8 * half];
      af[ms].u[1] = *(const uint4*)&ldsA[buf][r * 32 + 16 + 8 * half];
    }
    for (int ns = 0; ns < 2; ++ns) {
      int n = wn * 32 + ns * 16 + lane16;        // B layout: K = 8*half + e
      bf[ns].u[0] = *(const uint4*)&ldsBt[buf][n * 32 + 8 * half];
      bf[ns].u[1] = *(const uint4*)&ldsBt[buf][n * 32 + 16 + 8 * half];
    }
    for (int ms = 0; ms < 4; ++ms)
      for (int ns = 0; ns < 2; ++ns)
        acc[ms][ns] = __builtin_amdgcn_wmma_f32_16x16x32_bf16(
            false, af[ms].v, false, bf[ns].v, (short)0, acc[ms][ns], false, false);
  };

  // software pipeline: one barrier per K-step, copy of tile k+1 overlaps
  // the WMMAs of tile k.
  loadB(0);
  issueA(0, 0);
  commit(0);
  __syncthreads();
  const int NT = K / 32;
  for (int kt = 0; kt < NT; ++kt) {
    const int cur = kt & 1;
    if (kt + 1 < NT) { loadB((kt + 1) * 32); issueA(cur ^ 1, (kt + 1) * 32); }
    compute(cur);
    if (kt + 1 < NT) commit(cur ^ 1);
    __syncthreads();
  }

  for (int ms = 0; ms < 4; ++ms)
    for (int ns = 0; ns < 2; ++ns)
      for (int v = 0; v < 8; ++v) {
        int row = m0 + wm * 64 + ms * 16 + v + 8 * half;
        int col = n0 + wn * 32 + ns * 16 + lane16;
        float val = acc[ms][ns][v];
        if constexpr (OUT_BF16)
          ((unsigned short*)Cptr)[(size_t)row * N + col] = f2bf(val);
        else
          ((float*)Cptr)[(size_t)row * N + col] = val;
      }
}

// ---------------------------------------------------------------------------
// Kernel 5: block-sparse flash attention.  One (b, h, 64-row query block)
// per workgroup of 4 waves; each wave owns 16 query rows.  K chunk staged in
// LDS once per WG (async when available; row-major == B-operand staging).
// V chunk staged transposed.  Mask applied from packed 64-bit bitmaps.
// ---------------------------------------------------------------------------
__global__ __launch_bounds__(128) void bb_attention(
    const unsigned short* __restrict__ Qb, const unsigned short* __restrict__ Kb,
    const unsigned short* __restrict__ Vb,
    const unsigned long long* __restrict__ maskBits,
    const unsigned char* __restrict__ blockAny, unsigned short* __restrict__ Ob) {
  __shared__ unsigned short ldsK[BLKSZ * DHEAD];    // [key][d], stride 128
  __shared__ unsigned short ldsVt[DHEAD * BLKSZ];   // [d][key], stride 64
  __shared__ unsigned short ldsP[4 * 16 * BLKSZ];   // per-wave P (16 x 64)

  const int bid = blockIdx.x;
  const int qb = bid & 31;
  const int h = (bid >> 5) & 15;
  const int b = bid >> 9;
  const int tid = threadIdx.x;
  const int wid = tid >> 5, lane = tid & 31, half = lane >> 4, lane16 = lane & 15;

  const size_t rstr = (size_t)NHEAD * DHEAD;        // 2048 elems per token
  const size_t hoff = (size_t)h * DHEAD;
  const int q0 = qb * BLKSZ + wid * 16;

  // ---- Q fragments for this wave's 16 rows (4 A-operands over DH=128) ----
  const unsigned short* Qbase = Qb + ((size_t)b * S_LEN + q0) * rstr + hoff;
  FragA qf[4];
  for (int ds = 0; ds < 4; ++ds) {
    const unsigned short* p = Qbase + (size_t)lane16 * rstr + ds * 32 + 8 * half;
    qf[ds].u[0] = *(const uint4*)p;
    qf[ds].u[1] = *(const uint4*)(p + 16);
  }

  float rowM[8], rowL[8];
  v8f acc[8];
  for (int v = 0; v < 8; ++v) { rowM[v] = NEGBIG; rowL[v] = 0.f; }
  for (int dt = 0; dt < 8; ++dt)
    for (int v = 0; v < 8; ++v) acc[dt][v] = 0.f;

  const unsigned char* liveRow = blockAny + qb * NBLK;
  unsigned short* Pw = ldsP + wid * (16 * BLKSZ);

  for (int kb = 0; kb < NBLK; ++kb) {
    if (!liveRow[kb]) continue;                     // skip dead key block
    const int kc = kb * BLKSZ;
    __syncthreads();                                // LDS buffers free

    // ---- stage K chunk (64 x 128, row-major) -- async if available ----
    for (int i = 0; i < 8; ++i) {
      int g = tid + i * 128;                        // 1024 chunks of 8 bf16
      int kk = g >> 4, dof = (g & 15) * 8;
      const unsigned short* gp =
          &Kb[((size_t)b * S_LEN + kc + kk) * rstr + hoff + dof];
#if BB_ASYNC
      __builtin_amdgcn_global_load_async_to_lds_b128(
          AS1P(gp), AS3P(&ldsK[kk * DHEAD + dof]), 0, 0);
#else
      *(uint4*)&ldsK[kk * DHEAD + dof] = *(const uint4*)gp;
#endif
    }
    // ---- stage V chunk transposed (manual scatter) ----
    for (int i = 0; i < 8; ++i) {
      int g = tid + i * 128;
      int kk = g >> 4, dof = (g & 15) * 8;
      uint4 vv = *(const uint4*)&Vb[((size_t)b * S_LEN + kc + kk) * rstr + hoff + dof];
      const unsigned short* sv = (const unsigned short*)&vv;
      for (int j = 0; j < 8; ++j) ldsVt[(dof + j) * BLKSZ + kk] = sv[j];
    }
#if BB_ASYNC
    __builtin_amdgcn_s_wait_asynccnt(0);
#endif
    __syncthreads();

    // ---- packed mask words for this lane's 8 rows ----
    unsigned long long mrow[8];
    for (int v = 0; v < 8; ++v)
      mrow[v] = maskBits[(size_t)(q0 + v + 8 * half) * NBLK + kb];

    // ---- scores: 4 tiles of 16x16 from LDS-staged K ----
    v8f sc[4];
    for (int t = 0; t < 4; ++t) {
      v8f c;
      for (int i = 0; i < 8; ++i) c[i] = 0.f;
      for (int ds = 0; ds < 4; ++ds) {
        FragA kf;
        const unsigned short* p = &ldsK[(t * 16 + lane16) * DHEAD + ds * 32 + 8 * half];
        kf.u[0] = *(const uint4*)p;
        kf.u[1] = *(const uint4*)(p + 16);
        c = __builtin_amdgcn_wmma_f32_16x16x32_bf16(false, qf[ds].v, false, kf.v,
                                                    (short)0, c, false, false);
      }
      sc[t] = c;
    }

    // ---- scale + bitmap mask ----
    for (int t = 0; t < 4; ++t) {
      int cl = t * 16 + lane16;
      for (int v = 0; v < 8; ++v)
        sc[t][v] = ((mrow[v] >> cl) & 1ull) ? sc[t][v] * QKSCALE : NEGBIG;
    }

    // ---- online softmax (row reduce across 16-lane halves) ----
    for (int v = 0; v < 8; ++v) {
      float mc = fmaxf(fmaxf(sc[0][v], sc[1][v]), fmaxf(sc[2][v], sc[3][v]));
      for (int off = 1; off < 16; off <<= 1) mc = fmaxf(mc, __shfl_xor(mc, off, 32));
      float newM = fmaxf(rowM[v], mc);
      float alpha = __expf(rowM[v] - newM);
      rowM[v] = newM;
      float ps = 0.f;
      for (int t = 0; t < 4; ++t) {
        float p = __expf(sc[t][v] - newM);          // masked -> exp(-huge) = 0
        sc[t][v] = p;
        ps += p;
      }
      for (int off = 1; off < 16; off <<= 1) ps += __shfl_xor(ps, off, 32);
      rowL[v] = rowL[v] * alpha + ps;
      for (int dt = 0; dt < 8; ++dt) acc[dt][v] *= alpha;
    }

    // ---- P (C-layout) -> LDS -> A-operand layout as bf16 ----
    for (int t = 0; t < 4; ++t)
      for (int v = 0; v < 8; ++v)
        Pw[(v + 8 * half) * BLKSZ + t * 16 + lane16] = f2bf(sc[t][v]);
    __syncthreads();

    // ---- O += P @ V : 16 WMMAs (2 K-steps x 8 d-tiles) ----
    for (int ks = 0; ks < 2; ++ks) {
      FragA pf;
      const unsigned short* pp = Pw + lane16 * BLKSZ + ks * 32 + 8 * half;
      pf.u[0] = *(const uint4*)pp;
      pf.u[1] = *(const uint4*)(pp + 16);
      for (int dt = 0; dt < 8; ++dt) {
        FragA vf;
        const unsigned short* vp = ldsVt + (dt * 16 + lane16) * BLKSZ + ks * 32 + 8 * half;
        vf.u[0] = *(const uint4*)vp;
        vf.u[1] = *(const uint4*)(vp + 16);
        acc[dt] = __builtin_amdgcn_wmma_f32_16x16x32_bf16(false, pf.v, false, vf.v,
                                                          (short)0, acc[dt], false, false);
      }
    }
  }

  // ---- normalize and store bf16 (B,S,H,DH) for the output projection ----
  float inv[8];
  for (int v = 0; v < 8; ++v) inv[v] = 1.f / rowL[v];   // globals ensure L > 0
  for (int dt = 0; dt < 8; ++dt)
    for (int v = 0; v < 8; ++v) {
      int qrow = q0 + v + 8 * half;
      int d = dt * 16 + lane16;
      Ob[((size_t)b * S_LEN + qrow) * rstr + hoff + d] = f2bf(acc[dt][v] * inv[v]);
    }
}

// ---------------------------------------------------------------------------
// Launch: convert X -> mask pack/bitmap -> 3 QKV GEMMs -> attention -> out.
// ---------------------------------------------------------------------------
extern "C" void kernel_launch(void* const* d_in, const int* in_sizes, int n_in,
                              void* d_out, int out_size, void* d_ws, size_t ws_size,
                              hipStream_t stream) {
  (void)in_sizes; (void)n_in; (void)out_size; (void)ws_size;
  const float* X  = (const float*)d_in[0];
  const float* Wq = (const float*)d_in[1];
  const float* Wk = (const float*)d_in[2];
  const float* Wv = (const float*)d_in[3];
  const float* Wo = (const float*)d_in[4];
  const unsigned char* mask = (const unsigned char*)d_in[5];
  float* out = (float*)d_out;

  char* ws = (char*)d_ws;
  const size_t bufB = (size_t)NBATCH * S_LEN * NHEAD * DHEAD * sizeof(unsigned short);
  unsigned short* Xb = (unsigned short*)(ws);
  unsigned short* Qb = (unsigned short*)(ws + bufB);
  unsigned short* Kb = (unsigned short*)(ws + 2 * bufB);
  unsigned short* Vb = (unsigned short*)(ws + 3 * bufB);
  unsigned short* Ab = (unsigned short*)(ws + 4 * bufB);
  unsigned long long* maskBits = (unsigned long long*)(ws + 5 * bufB);
  unsigned char* blockAny = (unsigned char*)(ws + 5 * bufB + (size_t)S_LEN * NBLK * 8);

  const int M = NBATCH * S_LEN;            // 4096
  const int N = NHEAD * DHEAD;             // 2048
  const int K = HIDDEN;                    // 2048

  bb_convert<<<(M * K / 4 + 255) / 256, 256, 0, stream>>>(X, Xb, M * K / 4);
  bb_mask_pack<<<(S_LEN * NBLK) / 256, 256, 0, stream>>>(mask, maskBits);
  bb_mask_blocks<<<NBLK * NBLK, 64, 0, stream>>>(maskBits, blockAny);

  dim3 ggrid(N / 128, M / 128);            // (16, 32)
  bb_gemm<true><<<ggrid, 256, 0, stream>>>(Xb, Wq, Qb, M, N, K);
  bb_gemm<true><<<ggrid, 256, 0, stream>>>(Xb, Wk, Kb, M, N, K);
  bb_gemm<true><<<ggrid, 256, 0, stream>>>(Xb, Wv, Vb, M, N, K);

  bb_attention<<<NBATCH * NHEAD * NBLK, 128, 0, stream>>>(Qb, Kb, Vb, maskBits,
                                                          blockAny, Ab);

  dim3 ogrid(HIDDEN / 128, M / 128);
  bb_gemm<false><<<ogrid, 256, 0, stream>>>(Ab, Wo, out, M, HIDDEN, N);
}
